// MultiHeadAttention_80204219286041
// MI455X (gfx1250) — matrix operand order repaired and checked by reading, not compiled, not run on hardware
//
#include <hip/hip_runtime.h>
#include <hip/hip_bf16.h>
#include <math.h>

// Problem constants (match reference)
constexpr int nB = 8, nL = 1024, nD = 512, nH = 8, nDK = 64, nE = 8;
constexpr float LN_EPS = 1e-5f;
constexpr float QSCALE = 0.04419417382415922f; // 1/sqrt(512)

typedef __attribute__((ext_vector_type(16))) __bf16 v16bf;
typedef __attribute__((ext_vector_type(8)))  __bf16 v8bf;
typedef __attribute__((ext_vector_type(8)))  float  v8f;
typedef __attribute__((ext_vector_type(4)))  float  v4f;

__device__ __forceinline__ v16bf cat8(v8bf lo, v8bf hi) {
  return __builtin_shufflevector(lo, hi, 0, 1, 2, 3, 4, 5, 6, 7,
                                 8, 9, 10, 11, 12, 13, 14, 15);
}

// ---------------- WMMA fragment loaders (CDNA5 16x16x32 bf16 layouts) --------
// A 16x32 (MxK): lanes 0-15 row M=lane, K {0..7,16..23}; lanes 16-31 row
// M=lane-16, K {8..15,24..31}. Two 16-byte contiguous loads per lane.
__device__ __forceinline__ v16bf ld_a_bf(const __bf16* base, int ld, int lane) {
  const int row = lane & 15, kh = (lane >> 4) & 1;
  const __bf16* p = base + (size_t)row * ld + kh * 8;
  v8bf lo = *(const v8bf*)(p);
  v8bf hi = *(const v8bf*)(p + 16);
  return cat8(lo, hi);
}
// Same, f32 source, convert in-register (used for softmax probs).
__device__ __forceinline__ v16bf ld_a_f32(const float* base, int ld, int lane) {
  const int row = lane & 15, kh = (lane >> 4) & 1;
  const float* p = base + (size_t)row * ld + kh * 8;
  v4f f0 = *(const v4f*)(p);
  v4f f1 = *(const v4f*)(p + 4);
  v4f f2 = *(const v4f*)(p + 16);
  v4f f3 = *(const v4f*)(p + 20);
  v16bf a;
#pragma unroll
  for (int t = 0; t < 4; ++t) {
    a[t]      = (__bf16)f0[t];
    a[4 + t]  = (__bf16)f1[t];
    a[8 + t]  = (__bf16)f2[t];
    a[12 + t] = (__bf16)f3[t];
  }
  return a;
}
// B 32x16 (KxN) from an N-major source (row = n, k contiguous): lanes 0-15
// hold K {0..7,16..23} of column N=lane; lanes 16-31 hold K {8..15,24..31}. Two b128 loads.
__device__ __forceinline__ v16bf ld_b_nmaj(const __bf16* base, int ld, int lane) {
  const int col = lane & 15, kh = (lane >> 4) & 1;
  const __bf16* p = base + (size_t)col * ld + kh * 8;
  v8bf lo = *(const v8bf*)(p);
  v8bf hi = *(const v8bf*)(p + 16);
  return cat8(lo, hi);
}

#define WMMA_BF16(A, Bf, C) \
  __builtin_amdgcn_wmma_f32_16x16x32_bf16(false, (A), false, (Bf), (short)0, (C), false, false)

// ---------------- 1: pooled mean over L --------------------------------------
__global__ void pool_kernel(const float* __restrict__ x, float* __restrict__ pooled) {
  int idx = blockIdx.x * blockDim.x + threadIdx.x; // b*nD + d
  if (idx >= nB * nD) return;
  int b = idx / nD, d = idx % nD;
  const float* p = x + (size_t)b * nL * nD + d;
  float s = 0.f;
  for (int l = 0; l < nL; ++l) s += p[(size_t)l * nD];
  pooled[idx] = s * (1.0f / nL);
}

// ---------------- 2: gate MLP (relu fc1, sigmoid fc2) ------------------------
__global__ void gate_kernel(const float* __restrict__ pooled,
                            const float* __restrict__ fc1_w, const float* __restrict__ fc1_b,
                            const float* __restrict__ fc2_w, const float* __restrict__ fc2_b,
                            float* __restrict__ gate) {
  __shared__ float hsh[nB * nE];
  int t = threadIdx.x;         // 0..63
  int b = t / nE, e = t % nE;
  float s = fc1_b[e];
  for (int d = 0; d < nD; ++d) s += pooled[b * nD + d] * fc1_w[e * nD + d];
  hsh[t] = fmaxf(s, 0.0f);
  __syncthreads();
  float g = fc2_b[e];
  for (int j = 0; j < nE; ++j) g += hsh[b * nE + j] * fc2_w[e * nE + j];
  gate[t] = 1.0f / (1.0f + expf(-g));
}

// ---------------- 3: expert-mixed weights -> bf16, stored TRANSPOSED ---------
// wmixT[z][n][k] = sum_e gate[b,e] * W[e][k][n]   (z = which*nB + b)
// so GEMM B-fragments are k-contiguous 16B loads.
__global__ void wmix_kernel(const float* __restrict__ Wq, const float* __restrict__ Wk,
                            const float* __restrict__ Wv, const float* __restrict__ gate,
                            __bf16* __restrict__ wmixT) {
  const size_t per = (size_t)nB * nD * 512;
  size_t idx = (size_t)blockIdx.x * blockDim.x + threadIdx.x;
  if (idx >= 3 * per) return;
  int which = (int)(idx / per);
  size_t r = idx % per;
  int b = (int)(r / ((size_t)nD * 512));
  int dh = (int)(r % ((size_t)nD * 512));
  int d = dh >> 9;          // k index (model dim)
  int h = dh & 511;         // n index (head*dk), contiguous across threads -> coalesced reads
  const float* W = which == 0 ? Wq : which == 1 ? Wk : Wv;
  float s = 0.f;
#pragma unroll
  for (int e = 0; e < nE; ++e) s += gate[b * nE + e] * W[(size_t)e * nD * 512 + dh];
  wmixT[((size_t)which * nB + b) * nD * 512 + (size_t)h * 512 + d] = (__bf16)s;
}

__global__ void cvt_wo_kernel(const float* __restrict__ wo, __bf16* __restrict__ wo_bf) {
  int i = blockIdx.x * blockDim.x + threadIdx.x;
  if (i < nD * nH * nDK) wo_bf[i] = (__bf16)wo[i];
}

// ---------------- 4: q/k/v projection GEMM -----------------------------------
// C[1024x512] = A_f32[1024x512] * WmixT; grid (N/64, M/128, 3*B), block 128.
// Wave = 32 rows x 64 cols = 2 A-frags x 4 B-frags = 8 WMMA / K-step. No LDS.
// qs scaled by 1/sqrt(512); V written transposed (vT[bh][dv][l]).
// Epilogue is branchless: dst pointer + strides + scale are block-uniform.
__global__ void proj_gemm_kernel(const float* __restrict__ q, const float* __restrict__ k,
                                 const float* __restrict__ v, const __bf16* __restrict__ wmixT,
                                 __bf16* __restrict__ qs, __bf16* __restrict__ ks,
                                 __bf16* __restrict__ vT) {
  const int n0 = blockIdx.x * 64;
  const int m0 = blockIdx.y * 128;
  const int z  = blockIdx.z;                  // which*nB + b
  const int which = z / nB, b = z % nB;
  const float* A = (which == 0 ? q : which == 1 ? k : v) + (size_t)b * nL * nD;
  const __bf16* Bt = wmixT + (size_t)z * nD * 512;
  const int lane = threadIdx.x & 31, wid = threadIdx.x >> 5;
  const int mw = m0 + wid * 32;
  v8f zac = {};
  v8f acc[2][4] = {{zac, zac, zac, zac}, {zac, zac, zac, zac}};
  for (int kb = 0; kb < nD; kb += 32) {
    v16bf a0 = ld_a_f32(A + (size_t)mw * nD + kb, nD, lane);
    v16bf a1 = ld_a_f32(A + (size_t)(mw + 16) * nD + kb, nD, lane);
#pragma unroll
    for (int tn = 0; tn < 4; ++tn) {
      v16bf bf = ld_b_nmaj(Bt + (size_t)(n0 + tn * 16) * 512 + kb, 512, lane);
      acc[0][tn] = WMMA_BF16(a0, bf, acc[0][tn]);
      acc[1][tn] = WMMA_BF16(a1, bf, acc[1][tn]);
    }
  }
  // Block-uniform destination selection (no per-element control flow).
  // q/k layout: [bh][l][dk]  -> off = h*(L*64) + row*64 + d
  // vT  layout: [bh][dv][l]  -> off = h*(64*L) + row*1  + d*L
  __bf16* dst = (which == 0) ? qs : (which == 1) ? ks : vT;
  dst += (size_t)b * nH * nL * 64;
  const size_t rowStride = (which == 2) ? 1 : 64;
  const size_t colStride = (which == 2) ? (size_t)nL : 1;
  const float scale = (which == 0) ? QSCALE : 1.0f;
  const int rofs = (lane >> 4) * 8, col0 = lane & 15;
#pragma unroll
  for (int am = 0; am < 2; ++am)
#pragma unroll
    for (int tn = 0; tn < 4; ++tn)
#pragma unroll
      for (int r = 0; r < 8; ++r) {
        int row = mw + am * 16 + rofs + r;
        int col = n0 + tn * 16 + col0;
        int h = col >> 6, d = col & 63;
        dst[(size_t)h * (nL * 64) + (size_t)row * rowStride + (size_t)d * colStride] =
            (__bf16)(acc[am][tn][r] * scale);
      }
}

// ---------------- 5: scores S = Qs * Ks^T -> f32 into d_out attn region ------
// grid (L/64, L/128, B*H), block 128. Scale folded into Qs.
__global__ void scores_kernel(const __bf16* __restrict__ qs, const __bf16* __restrict__ ks,
                              float* __restrict__ attn) {
  const int n0 = blockIdx.x * 64;
  const int m0 = blockIdx.y * 128;
  const int z = blockIdx.z;                   // b*nH + h
  const int b = z >> 3, h = z & 7;
  const __bf16* Q = qs + (size_t)z * nL * 64;
  const __bf16* K = ks + (size_t)z * nL * 64;
  const int lane = threadIdx.x & 31, wid = threadIdx.x >> 5;
  const int mw = m0 + wid * 32;
  v8f zac = {};
  v8f acc[2][4] = {{zac, zac, zac, zac}, {zac, zac, zac, zac}};
#pragma unroll
  for (int kb = 0; kb < 64; kb += 32) {
    v16bf a0 = ld_a_bf(Q + (size_t)mw * 64 + kb, 64, lane);
    v16bf a1 = ld_a_bf(Q + (size_t)(mw + 16) * 64 + kb, 64, lane);
#pragma unroll
    for (int tn = 0; tn < 4; ++tn) {
      v16bf bf = ld_b_nmaj(K + (size_t)(n0 + tn * 16) * 64 + kb, 64, lane);
      acc[0][tn] = WMMA_BF16(a0, bf, acc[0][tn]);
      acc[1][tn] = WMMA_BF16(a1, bf, acc[1][tn]);
    }
  }
  float* out = attn + ((size_t)(h * nB + b) * nL) * nL;   // torch layout [H*B, L, L]
  const int rofs = (lane >> 4) * 8, col0 = lane & 15;
#pragma unroll
  for (int am = 0; am < 2; ++am)
#pragma unroll
    for (int tn = 0; tn < 4; ++tn)
#pragma unroll
      for (int r = 0; r < 8; ++r)
        out[(size_t)(mw + am * 16 + rofs + r) * nL + n0 + tn * 16 + col0] = acc[am][tn][r];
}

// ---------------- 6: row softmax in place (d_out attn region) ----------------
__global__ void softmax_kernel(float* __restrict__ attn) {
  const size_t row = blockIdx.x;              // 0 .. B*H*L-1
  float* p = attn + row * (size_t)nL;
  const int tid = threadIdx.x, lane = tid & 31, wid = tid >> 5;
  __shared__ float sm[8];
  __shared__ float ssum[8];
  v4f x = *(const v4f*)(p + tid * 4);
  float m = fmaxf(fmaxf(x[0], x[1]), fmaxf(x[2], x[3]));
  for (int off = 16; off > 0; off >>= 1) m = fmaxf(m, __shfl_xor(m, off, 32));
  if (lane == 0) sm[wid] = m;
  __syncthreads();
  m = sm[0];
#pragma unroll
  for (int i = 1; i < 8; ++i) m = fmaxf(m, sm[i]);
  float s = 0.f;
#pragma unroll
  for (int i = 0; i < 4; ++i) { x[i] = __expf(x[i] - m); s += x[i]; }
  for (int off = 16; off > 0; off >>= 1) s += __shfl_xor(s, off, 32);
  if (lane == 0) ssum[wid] = s;
  __syncthreads();
  s = 0.f;
#pragma unroll
  for (int i = 0; i < 8; ++i) s += ssum[i];
  const float inv = 1.0f / s;
#pragma unroll
  for (int i = 0; i < 4; ++i) x[i] *= inv;
  *(v4f*)(p + tid * 4) = x;
}

// ---------------- 7: head = P[1024x1024] * V[1024x64] ------------------------
// grid (L/128, B*H), block 128. P read f32 from d_out (converted in-register),
// V read from the transposed vT buffer (k-contiguous B fragments). No LDS.
__global__ void head_gemm_kernel(const float* __restrict__ attn, const __bf16* __restrict__ vT,
                                 __bf16* __restrict__ headb) {
  const int m0 = blockIdx.x * 128;
  const int z = blockIdx.y;                   // b*nH + h
  const int b = z >> 3, h = z & 7;
  const float*  P = attn + ((size_t)(h * nB + b) * nL) * nL;
  const __bf16* V = vT + (size_t)z * 64 * nL;
  const int lane = threadIdx.x & 31, wid = threadIdx.x >> 5;
  const int mw = m0 + wid * 32;
  v8f zac = {};
  v8f acc[2][4] = {{zac, zac, zac, zac}, {zac, zac, zac, zac}};
  for (int kb = 0; kb < nL; kb += 32) {
    v16bf a0 = ld_a_f32(P + (size_t)mw * nL + kb, nL, lane);
    v16bf a1 = ld_a_f32(P + (size_t)(mw + 16) * nL + kb, nL, lane);
#pragma unroll
    for (int tn = 0; tn < 4; ++tn) {
      v16bf bf = ld_b_nmaj(V + (size_t)(tn * 16) * nL + kb, nL, lane);
      acc[0][tn] = WMMA_BF16(a0, bf, acc[0][tn]);
      acc[1][tn] = WMMA_BF16(a1, bf, acc[1][tn]);
    }
  }
  const int rofs = (lane >> 4) * 8, col0 = lane & 15;
#pragma unroll
  for (int am = 0; am < 2; ++am)
#pragma unroll
    for (int tn = 0; tn < 4; ++tn)
#pragma unroll
      for (int r = 0; r < 8; ++r) {
        int row = mw + am * 16 + rofs + r;
        int col = tn * 16 + col0;             // dv index, N == 64 total
        headb[((size_t)b * nL + row) * 512 + h * 64 + col] = (__bf16)acc[am][tn][r];
      }
}

// ---------------- 8: out = head * wo_w^T + wo_b + q (pre-LN, f32) ------------
// grid (D/64, L/128, B), block 128. wo_w rows are n-major (k-contiguous). No LDS.
__global__ void outproj_kernel(const __bf16* __restrict__ headb, const __bf16* __restrict__ wo_bf,
                               const float* __restrict__ wo_b, const float* __restrict__ q,
                               float* __restrict__ pre) {
  const int n0 = blockIdx.x * 64;
  const int m0 = blockIdx.y * 128;
  const int b  = blockIdx.z;
  const __bf16* A = headb + (size_t)b * nL * 512;
  const int lane = threadIdx.x & 31, wid = threadIdx.x >> 5;
  const int mw = m0 + wid * 32;
  v8f zac = {};
  v8f acc[2][4] = {{zac, zac, zac, zac}, {zac, zac, zac, zac}};
  for (int kb = 0; kb < 512; kb += 32) {
    v16bf a0 = ld_a_bf(A + (size_t)mw * 512 + kb, 512, lane);
    v16bf a1 = ld_a_bf(A + (size_t)(mw + 16) * 512 + kb, 512, lane);
#pragma unroll
    for (int tn = 0; tn < 4; ++tn) {
      // out[m,d] = sum_k head[m,k]*wo_w[d,k] -> B[k][n] = wo_w[n][k], k-contiguous
      v16bf bf = ld_b_nmaj(wo_bf + (size_t)(n0 + tn * 16) * 512 + kb, 512, lane);
      acc[0][tn] = WMMA_BF16(a0, bf, acc[0][tn]);
      acc[1][tn] = WMMA_BF16(a1, bf, acc[1][tn]);
    }
  }
  const int rofs = (lane >> 4) * 8, col0 = lane & 15;
#pragma unroll
  for (int am = 0; am < 2; ++am)
#pragma unroll
    for (int tn = 0; tn < 4; ++tn)
#pragma unroll
      for (int r = 0; r < 8; ++r) {
        int row = mw + am * 16 + rofs + r;
        int col = n0 + tn * 16 + col0;
        size_t o = ((size_t)b * nL + row) * 512 + col;
        pre[o] = acc[am][tn][r] + wo_b[col] + q[o];
      }
}

// ---------------- 9: layernorm -> d_out out region ---------------------------
__global__ void ln_kernel(const float* __restrict__ pre, const float* __restrict__ ln_g,
                          const float* __restrict__ ln_b, float* __restrict__ out) {
  const size_t row = blockIdx.x;              // 0 .. B*L-1
  const float* p = pre + row * nD;
  const int tid = threadIdx.x, lane = tid & 31, wid = tid >> 5; // 128 threads
  __shared__ float shs[4];
  __shared__ float shq[4];
  v4f x = *(const v4f*)(p + tid * 4);
  float s = x[0] + x[1] + x[2] + x[3];
  float sq = x[0] * x[0] + x[1] * x[1] + x[2] * x[2] + x[3] * x[3];
  for (int off = 16; off > 0; off >>= 1) {
    s  += __shfl_xor(s,  off, 32);
    sq += __shfl_xor(sq, off, 32);
  }
  if (lane == 0) { shs[wid] = s; shq[wid] = sq; }
  __syncthreads();
  s = shs[0] + shs[1] + shs[2] + shs[3];
  sq = shq[0] + shq[1] + shq[2] + shq[3];
  const float mu  = s * (1.0f / nD);
  const float var = sq * (1.0f / nD) - mu * mu;
  const float inv = rsqrtf(var + LN_EPS);
  v4f g = *(const v4f*)(ln_g + tid * 4);
  v4f bb = *(const v4f*)(ln_b + tid * 4);
  v4f o;
#pragma unroll
  for (int i = 0; i < 4; ++i) o[i] = (x[i] - mu) * inv * g[i] + bb[i];
  *(v4f*)(out + row * nD + tid * 4) = o;
}

// =============================================================================
extern "C" void kernel_launch(void* const* d_in, const int* in_sizes, int n_in,
                              void* d_out, int out_size, void* d_ws, size_t ws_size,
                              hipStream_t stream) {
  (void)in_sizes; (void)n_in; (void)out_size; (void)ws_size;
  const float* q     = (const float*)d_in[0];
  const float* k     = (const float*)d_in[1];
  const float* v     = (const float*)d_in[2];
  const float* x_    = (const float*)d_in[3];
  const float* Wq    = (const float*)d_in[4];
  const float* Wk    = (const float*)d_in[5];
  const float* Wv    = (const float*)d_in[6];
  const float* fc1_w = (const float*)d_in[7];
  const float* fc1_b = (const float*)d_in[8];
  const float* fc2_w = (const float*)d_in[9];
  const float* fc2_b = (const float*)d_in[10];
  const float* wo_w  = (const float*)d_in[11];
  const float* wo_b  = (const float*)d_in[12];
  const float* ln_g  = (const float*)d_in[13];
  const float* ln_b  = (const float*)d_in[14];

  // Workspace layout (256B aligned)
  char* ws = (char*)d_ws;
  size_t off = 0;
  auto wsalloc = [&](size_t bytes) -> void* {
    void* p = ws + off;
    off += (bytes + 255) & ~(size_t)255;
    return p;
  };
  float*  pooled = (float*)wsalloc((size_t)nB * nD * 4);
  float*  gate   = (float*)wsalloc((size_t)nB * nE * 4);
  __bf16* wmixT  = (__bf16*)wsalloc((size_t)3 * nB * nD * 512 * 2);    // 12.6 MB
  __bf16* qsb    = (__bf16*)wsalloc((size_t)nB * nH * nL * 64 * 2);    // 8.4 MB
  __bf16* ksb    = (__bf16*)wsalloc((size_t)nB * nH * nL * 64 * 2);    // 8.4 MB
  __bf16* vTb    = (__bf16*)wsalloc((size_t)nB * nH * 64 * nL * 2);    // 8.4 MB
  __bf16* headb  = (__bf16*)wsalloc((size_t)nB * nL * 512 * 2);        // 8.4 MB
  __bf16* wo_bf  = (__bf16*)wsalloc((size_t)nD * nH * nDK * 2);        // 0.5 MB
  float*  pre    = (float*)wsalloc((size_t)nB * nL * nD * 4);          // 16.8 MB

  float* out_p  = (float*)d_out;                       // [B,L,D]
  float* attn_p = out_p + (size_t)nB * nL * nD;        // [H*B,L,L]

  // 1-3: gate path + weight mixing / conversion
  pool_kernel<<<(nB * nD + 255) / 256, 256, 0, stream>>>(x_, pooled);
  gate_kernel<<<1, 64, 0, stream>>>(pooled, fc1_w, fc1_b, fc2_w, fc2_b, gate);
  {
    size_t tot = (size_t)3 * nB * nD * 512;
    wmix_kernel<<<(unsigned)((tot + 255) / 256), 256, 0, stream>>>(Wq, Wk, Wv, gate, wmixT);
  }
  cvt_wo_kernel<<<(nD * nH * nDK + 255) / 256, 256, 0, stream>>>(wo_w, wo_bf);

  // 4: q/k/v projections (WMMA bf16, LDS-free)
  proj_gemm_kernel<<<dim3(512 / 64, nL / 128, 3 * nB), 128, 0, stream>>>(
      q, k, v, wmixT, qsb, ksb, vTb);

  // 5: attention scores (WMMA bf16) -> d_out attn region
  scores_kernel<<<dim3(nL / 64, nL / 128, nB * nH), 128, 0, stream>>>(qsb, ksb, attn_p);

  // 6: softmax in place
  softmax_kernel<<<(unsigned)((size_t)nB * nH * nL), 256, 0, stream>>>(attn_p);

  // 7: head = P @ V (WMMA bf16, P converted from f32 probs in-flight)
  head_gemm_kernel<<<dim3(nL / 128, nB * nH), 128, 0, stream>>>(attn_p, vTb, headb);

  // 8: output projection + bias + residual
  outproj_kernel<<<dim3(nD / 64, nL / 128, nB), 128, 0, stream>>>(headb, wo_bf, wo_b, q, pre);

  // 9: layernorm -> d_out out region
  ln_kernel<<<(unsigned)((size_t)nB * nL), 128, 0, stream>>>(pre, ln_g, ln_b, out_p);
}
